// MultiQueryAttention_72267119722507
// MI455X (gfx1250) — matrix operand order repaired and checked by reading, not compiled, not run on hardware
//
#include <hip/hip_runtime.h>
#include <hip/hip_bf16.h>
#include <math.h>

#define T_LEN 4096
#define EMBD  1024
#define NH    8
#define HD    128

typedef __attribute__((ext_vector_type(16))) __bf16 v16bf;
typedef __attribute__((ext_vector_type(8)))  float  v8f;

__device__ __forceinline__ unsigned short f2bf(float f) {
  union { float f; unsigned u; } c; c.f = f;
  unsigned r = (c.u + 0x7FFFu + ((c.u >> 16) & 1u)) >> 16;   // round-to-nearest-even
  return (unsigned short)r;
}

__device__ __forceinline__ unsigned pack2bf(float lo, float hi) {
  return (unsigned)f2bf(lo) | ((unsigned)f2bf(hi) << 16);
}

// A-matrix fragment (16x32 bf16), documented CDNA5 layout:
// lane m = lane&15, half h = lane>>4; VGPR p holds K = (p>=4?16:0) + 8h + 2(p&3), +1
__device__ __forceinline__ v16bf load_fragA(const unsigned short* tile, int m, int ld, int h) {
  union { v16bf v; unsigned u[8]; } f;
  const unsigned short* src = tile + m * ld;
#pragma unroll
  for (int p = 0; p < 8; ++p) {
    int k = ((p & 4) << 2) + (h << 3) + ((p & 3) << 1);
    f.u[p] = *(const unsigned*)(src + k);
  }
  return f.v;
}

// B-matrix fragment (32x16 bf16): lane n = lane&15, VGPR p holds K = (p>=4?16:0) + 8h + 2(p&3), +1
__device__ __forceinline__ v16bf load_fragB(const unsigned short* tile, int n, int ld, int koff, int h) {
  union { v16bf v; unsigned u[8]; } f;
  const unsigned short* src = tile + n * ld + koff + (h << 3);
#pragma unroll
  for (int p = 0; p < 8; ++p) f.u[p] = *(const unsigned*)(src + ((p & 4) << 2) + ((p & 3) << 1));
  return f.v;
}

__device__ __forceinline__ v8f wmma_bf16(v16bf a, v16bf b, v8f c) {
  return __builtin_amdgcn_wmma_f32_16x16x32_bf16(false, a, false, b, (short)0, c, false, false);
}

// sinusoidal PE: ln(10000)/64 = 0.14391157
__device__ __forceinline__ float pe_val(int t, int d) {
  float ang = (float)t * __expf(-0.14391157f * (float)(d >> 1));
  return (d & 1) ? __cosf(ang) : __sinf(ang);
}

// ---------------------------------------------------------------------------
// Generic bf16-WMMA GEMM:  Out[MxN] = A[MxK] * B[KxN] (+bias) (+PE)
// PE_MODE: 0=none, 1=pe[t, n%128] on all cols (Q), 2=pe[t, n] on cols<128 (K half of KV)
// ---------------------------------------------------------------------------
template<int PE_MODE, bool OUT_BF16>
__global__ __launch_bounds__(256) void gemm_bf16_wmma(
    const float* __restrict__ A, const float* __restrict__ B,
    const float* __restrict__ bias, void* __restrict__ Out,
    int M, int N, int K)
{
  __shared__ __align__(16) unsigned short As[128][34];   // row-major, padded
  __shared__ __align__(16) unsigned short Bt[128][34];   // transposed [n][k], padded
  const int tid = threadIdx.x;
  const int lane = tid & 31, wid = tid >> 5;
  const int l15 = lane & 15, h = lane >> 4;
  const int wm = wid >> 2, wn = wid & 3;                 // 2x4 wave grid, 64x32 per wave
  const int mb = blockIdx.y * 128, nb = blockIdx.x * 128;

  v8f acc[4][2] = {};

  for (int kb = 0; kb < K; kb += 32) {
    // warm L2->WGP path for the next K-step while this one computes
    if (kb + 32 < K) {
      __builtin_prefetch(&A[(size_t)(mb + (tid & 127)) * K + kb + 32 + ((tid >> 7) << 4)], 0, 1);
      __builtin_prefetch(&B[(size_t)(kb + 32 + (tid >> 3)) * N + nb + ((tid & 7) << 4)], 0, 1);
    }
    // A tile 128x32: b64 global loads, packed b32 LDS stores
#pragma unroll
    for (int r = 0; r < 8; ++r) {
      int idx = r * 256 + tid;
      int m = idx >> 4, k2 = (idx & 15) << 1;
      const float2 va = *(const float2*)(A + (size_t)(mb + m) * K + kb + k2);
      *(unsigned*)(&As[m][k2]) = pack2bf(va.x, va.y);
    }
    // B tile 32x128 -> transposed [n][k]; pair two K-rows, packed b32 store
#pragma unroll
    for (int r = 0; r < 8; ++r) {
      int idx = r * 256 + tid;
      int n = idx & 127, k2 = (idx >> 7) << 1;
      float b0 = B[(size_t)(kb + k2) * N + nb + n];
      float b1 = B[(size_t)(kb + k2 + 1) * N + nb + n];
      *(unsigned*)(&Bt[n][k2]) = pack2bf(b0, b1);
    }
    __syncthreads();

    v16bf af[4], bw[2];
#pragma unroll
    for (int mt = 0; mt < 4; ++mt) af[mt] = load_fragA(&As[0][0], wm * 64 + mt * 16 + l15, 34, h);
#pragma unroll
    for (int nt = 0; nt < 2; ++nt) bw[nt] = load_fragB(&Bt[0][0], wn * 32 + nt * 16 + l15, 34, 0, h);
#pragma unroll
    for (int mt = 0; mt < 4; ++mt)
#pragma unroll
      for (int nt = 0; nt < 2; ++nt)
        acc[mt][nt] = wmma_bf16(af[mt], bw[nt], acc[mt][nt]);
    __syncthreads();
  }

  // epilogue: C/D layout M = r + 8h, N = lane&15
#pragma unroll
  for (int mt = 0; mt < 4; ++mt)
#pragma unroll
    for (int nt = 0; nt < 2; ++nt)
#pragma unroll
      for (int r = 0; r < 8; ++r) {
        int mg = mb + wm * 64 + mt * 16 + r + h * 8;
        int ng = nb + wn * 32 + nt * 16 + l15;
        float v = acc[mt][nt][r];
        if (bias) v += bias[ng];
        if (PE_MODE == 1) v += pe_val(mg, ng & 127);
        if (PE_MODE == 2 && ng < 128) v += pe_val(mg, ng);
        if (OUT_BF16) ((unsigned short*)Out)[(size_t)mg * N + ng] = f2bf(v);
        else          ((float*)Out)[(size_t)mg * N + ng] = v;
      }
}

// ---------------------------------------------------------------------------
// Causal flash attention, MQA (shared K/V). Grid: (T/64, NH), 128 threads.
// Each wave owns 16 query rows; workgroup shares 32-key K/V blocks in LDS.
// ---------------------------------------------------------------------------
__global__ __launch_bounds__(128) void flash_mqa(
    const unsigned short* __restrict__ q,    // (T, NH*HD) bf16, PE already added
    const unsigned short* __restrict__ kv,   // (T, 2*HD)  bf16, PE on K half
    float* __restrict__ attn)                // (T, NH*HD) f32
{
  __shared__ __align__(16) unsigned short ksh[32][136];     // [key][d] row-major
  __shared__ __align__(16) unsigned short vsh[128][40];     // [d][key] transposed
  __shared__ __align__(16) unsigned short psh[4][16][40];   // per-wave P staging
  const int tid = threadIdx.x;
  const int lane = tid & 31, wid = tid >> 5;
  const int l15 = lane & 15, h = lane >> 4;
  const int hd = blockIdx.y;
  const int qbase = blockIdx.x * 64;
  const int qrow0 = qbase + wid * 16;
  const float scale = 0.08838834764831845f;  // 1/sqrt(128)

  // Q fragments for this wave's 16 rows (A-layout, loaded straight from global)
  v16bf qf[4];
#pragma unroll
  for (int s = 0; s < 4; ++s) {
    union { v16bf v; unsigned u[8]; } f;
    const unsigned short* src = q + (size_t)(qrow0 + l15) * (NH * HD) + hd * HD + s * 32;
#pragma unroll
    for (int p = 0; p < 8; ++p) {
      int k = ((p & 4) << 2) + (h << 3) + ((p & 3) << 1);
      f.u[p] = *(const unsigned*)(src + k);
    }
    qf[s] = f.v;
  }

  v8f o[8] = {};
  float mrun[8], lrun[8];
#pragma unroll
  for (int r = 0; r < 8; ++r) { mrun[r] = -3.0e38f; lrun[r] = 0.0f; }

  const int jmax = blockIdx.x * 2 + 2;          // causal: keys <= qbase+63
  for (int j = 0; j < jmax; ++j) {
    const int kbase = j * 32;
    if (j + 1 < jmax)  // prefetch next K/V block
      __builtin_prefetch(kv + (size_t)(kbase + 32 + (tid >> 2)) * (2 * HD) + ((tid & 3) << 6), 0, 1);
    // cooperative K/V load: 32 keys x 128 d each
#pragma unroll
    for (int r = 0; r < 16; ++r) {
      int idx = r * 128 + tid;
      int key = idx >> 6;
      int d = (idx & 63) * 2;
      const unsigned short* kp = kv + (size_t)(kbase + key) * (2 * HD);
      *(unsigned*)(&ksh[key][d]) = *(const unsigned*)(kp + d);
      unsigned uv = *(const unsigned*)(kp + HD + d);
      vsh[d][key]     = (unsigned short)(uv & 0xffffu);
      vsh[d + 1][key] = (unsigned short)(uv >> 16);
    }
    __syncthreads();

    // S = Q K^T  (two 16x16 key tiles, K-dim = d = 128)
    v8f s0 = {}, s1 = {};
#pragma unroll
    for (int s = 0; s < 4; ++s) {
      v16bf b0 = load_fragB(&ksh[0][0], l15,      136, s * 32, h);
      v16bf b1 = load_fragB(&ksh[0][0], 16 + l15, 136, s * 32, h);
      s0 = wmma_bf16(qf[s], b0, s0);
      s1 = wmma_bf16(qf[s], b1, s1);
    }

    // online softmax; row identity constant within each 16-lane half
#pragma unroll
    for (int r = 0; r < 8; ++r) {
      int qr  = qrow0 + r + h * 8;
      int kc0 = kbase + l15;
      int kc1 = kbase + 16 + l15;
      float x0 = (kc0 <= qr) ? s0[r] * scale : -3.0e38f;
      float x1 = (kc1 <= qr) ? s1[r] * scale : -3.0e38f;
      float mx = fmaxf(x0, x1);
#pragma unroll
      for (int off = 8; off >= 1; off >>= 1) mx = fmaxf(mx, __shfl_xor(mx, off, 16));
      float mn   = fmaxf(mrun[r], mx);
      float corr = __expf(mrun[r] - mn);
      float p0 = (kc0 <= qr) ? __expf(x0 - mn) : 0.0f;
      float p1 = (kc1 <= qr) ? __expf(x1 - mn) : 0.0f;
      float rs = p0 + p1;
#pragma unroll
      for (int off = 8; off >= 1; off >>= 1) rs += __shfl_xor(rs, off, 16);
      lrun[r] = lrun[r] * corr + rs;
      mrun[r] = mn;
#pragma unroll
      for (int dt = 0; dt < 8; ++dt) o[dt][r] *= corr;
      // D-layout -> A-layout bounce via per-wave LDS (intra-wave DS order, no barrier)
      psh[wid][r + h * 8][l15]      = f2bf(p0);
      psh[wid][r + h * 8][16 + l15] = f2bf(p1);
    }

    // O += P V
    v16bf pf = load_fragA(&psh[wid][0][0], l15, 40, h);
#pragma unroll
    for (int dt = 0; dt < 8; ++dt) {
      v16bf vf = load_fragB(&vsh[0][0], dt * 16 + l15, 40, 0, h);
      o[dt] = wmma_bf16(pf, vf, o[dt]);
    }
    __syncthreads();
  }

  // attn = O / l   (f32 to workspace; consumed by output projection GEMM)
#pragma unroll
  for (int dt = 0; dt < 8; ++dt)
#pragma unroll
    for (int r = 0; r < 8; ++r) {
      int row = qrow0 + r + h * 8;
      int col = hd * HD + dt * 16 + l15;
      attn[(size_t)row * (NH * HD) + col] = o[dt][r] / lrun[r];
    }
}

// ---------------------------------------------------------------------------
extern "C" void kernel_launch(void* const* d_in, const int* in_sizes, int n_in,
                              void* d_out, int out_size, void* d_ws, size_t ws_size,
                              hipStream_t stream) {
  (void)in_sizes; (void)n_in; (void)out_size; (void)ws_size;
  const float* x   = (const float*)d_in[0];
  const float* Wq  = (const float*)d_in[1];
  const float* bq  = (const float*)d_in[2];
  const float* Wkv = (const float*)d_in[3];
  const float* bkv = (const float*)d_in[4];
  const float* Wo  = (const float*)d_in[5];

  unsigned char* ws = (unsigned char*)d_ws;
  unsigned short* qb  = (unsigned short*)(ws);                       // 8 MB: q bf16 (T,1024)
  unsigned short* kvb = (unsigned short*)(ws + (size_t)(8u << 20));  // 2 MB: kv bf16 (T,256)
  float*          at  = (float*)(ws + (size_t)(10u << 20));          // 16 MB: attn f32 (T,1024)
  float* out = (float*)d_out;

  dim3 blk(256);
  // Q projection: q = x@Wq + bq + PE(t, n%128)   -> bf16
  gemm_bf16_wmma<1, true><<<dim3(EMBD / 128, T_LEN / 128), blk, 0, stream>>>(
      x, Wq, bq, qb, T_LEN, EMBD, EMBD);
  // KV projection: kv = x@Wkv + bkv, PE on first 128 cols (K) -> bf16
  gemm_bf16_wmma<2, true><<<dim3(2, T_LEN / 128), blk, 0, stream>>>(
      x, Wkv, bkv, kvb, T_LEN, 2 * HD, EMBD);
  // causal MQA flash attention -> attn f32
  flash_mqa<<<dim3(T_LEN / 64, NH), dim3(128), 0, stream>>>(qb, kvb, at);
  // output projection: out = attn @ Wo  -> f32
  gemm_bf16_wmma<0, false><<<dim3(EMBD / 128, T_LEN / 128), blk, 0, stream>>>(
      at, Wo, nullptr, out, T_LEN, EMBD, EMBD);
}
